// ViT_64398739636279
// MI455X (gfx1250) — compile-verified, matrix operand order checked by reading, not run on hardware
//
#include <hip/hip_runtime.h>
#include <hip/hip_bf16.h>
#include <stdint.h>

// ---------------------------------------------------------------------------
// CDNA5 wave32 WMMA fragments: v_wmma_f32_16x16x32_bf16, D = A(16x32) x B(32x16) + C
// A layout (ISA 7.12.2): lanes 0-15 row M=lane, K 0-7 in u[0..3], K 16-23 in u[4..7];
//                        lanes 16-31 row M=lane-16, K 8-15 / 24-31.
// B fragment symmetric column-striped -> build from K-contiguous storage.
// C/D: lane 0-15 col N=lane rows M=vgpr; lanes 16-31 rows M=8+vgpr.
// ---------------------------------------------------------------------------

typedef __attribute__((ext_vector_type(16))) __bf16 v16bf;
typedef __attribute__((ext_vector_type(8)))  float  v8f;
typedef unsigned int u32x4 __attribute__((ext_vector_type(4)));
typedef int          i32x8 __attribute__((ext_vector_type(8)));
typedef int          i32x4 __attribute__((ext_vector_type(4)));

union FragBF {
  v16bf    bf;
  unsigned u[8];
  uint4    q[2];
};

__device__ __forceinline__ unsigned pack_bf2(float a, float b) {
  unsigned short lo = __builtin_bit_cast(unsigned short, (__bf16)a);
  unsigned short hi = __builtin_bit_cast(unsigned short, (__bf16)b);
  return (unsigned)lo | ((unsigned)hi << 16);
}

__device__ __forceinline__ v8f wmma_bf16(const FragBF& a, const FragBF& b, v8f c) {
  return __builtin_amdgcn_wmma_f32_16x16x32_bf16(
      false, a.bf, false, b.bf, (short)0, c, false, false);
}

__device__ __forceinline__ void frag_load(FragBF& f, const __bf16* rowbase, int khalf) {
  f.q[0] = *(const uint4*)(rowbase + khalf);
  f.q[1] = *(const uint4*)(rowbase + 16 + khalf);
}

// ---------------------------------------------------------------------------
// Tensor Data Mover: 2-D tile load Global -> LDS (D# per ISA 08_async_tensor §8).
// dims/strides in data_size units. pad codes: interval 0=2dw..7=256dw, amount 0=1dw..
// Issued by one wave; caller pairs with s_wait_tensorcnt(0) + block barrier.
// (clang-23 toolchain: 6-arg builtin, extra i32x8 group zeroed for 2-D tiles.)
// ---------------------------------------------------------------------------
__device__ __forceinline__ void tdm_load_2d(
    unsigned lds_off, const void* gptr,
    unsigned tensor_d0, unsigned tensor_d1,
    unsigned tile_d0, unsigned tile_d1,
    unsigned stride0, unsigned data_size_code,
    unsigned pad_interval_code, unsigned pad_amount_code, bool pad_en)
{
  unsigned long long ga = (unsigned long long)(uintptr_t)gptr;
  u32x4 g0;
  g0[0] = 1u;                                            // count=1, user desc
  g0[1] = lds_off;                                       // lds_addr [63:32]
  g0[2] = (unsigned)(ga & 0xFFFFFFFFu);                  // global_addr[31:0]
  g0[3] = (unsigned)((ga >> 32) & 0x1FFFFFFu) | (2u << 30); // addr[56:32] | type=2

  i32x8 g1;
  g1[0] = (int)((data_size_code << 16) |
                ((pad_en ? 1u : 0u) << 20) |
                (pad_interval_code << 22) |
                (pad_amount_code << 25));                 // wg_mask=0, no atomic barrier
  g1[1] = (int)((tensor_d0 & 0xFFFFu) << 16);            // [63:48]=tensor_dim0 lo
  g1[2] = (int)((tensor_d0 >> 16) | ((tensor_d1 & 0xFFFFu) << 16));
  g1[3] = (int)((tensor_d1 >> 16) | ((tile_d0 & 0xFFFFu) << 16));
  g1[4] = (int)(tile_d1 & 0xFFFFu);                      // tile_dim2 = 0 (2-D)
  g1[5] = (int)stride0;                                  // dim0 stride [31:0]
  g1[6] = 0;                                             // stride hi, dim1 stride lo
  g1[7] = 0;

  i32x4 z4 = {0, 0, 0, 0};
  i32x8 z8 = {0, 0, 0, 0, 0, 0, 0, 0};
  __builtin_amdgcn_tensor_load_to_lds(g0, g1, z4, z4, z8, 0);
}

__device__ __forceinline__ unsigned lds_offset_of(const void* p) {
  return (unsigned)(uintptr_t)p;   // flat-LDS aperture: low 32 bits = LDS offset
}

// ---------------------------------------------------------------------------
// Kernel 1: QKV projection.  x[50176,768] (f32) @ w[768,2304] (f32) + b
// Output scattered as bf16 q/k/v [B*nh=3072][196][64].
// (fp32 sources need conversion, so staging stays manual vector loads.)
// ---------------------------------------------------------------------------
__global__ __launch_bounds__(256) void qkv_gemm_kernel(
    const float* __restrict__ x, const float* __restrict__ w,
    const float* __restrict__ bias,
    __bf16* __restrict__ qb, __bf16* __restrict__ kb, __bf16* __restrict__ vb)
{
  constexpr int K = 768, N = 2304, LDA = 40;   // 80B stride, 16B aligned
  __shared__ __bf16 As[128 * LDA];
  __shared__ __bf16 Bs[64 * LDA];              // transposed: [n][k]

  const int tid   = threadIdx.x;
  const int lane  = tid & 31;
  const int wave  = tid >> 5;
  const int l15   = lane & 15;
  const int khalf = (lane >> 4) << 3;
  const int rowoff = (lane >> 4) << 3;

  const int    n0 = blockIdx.x * 64;
  const size_t m0 = (size_t)blockIdx.y * 128;

  v8f acc[4] = {};

  for (int k0 = 0; k0 < K; k0 += 32) {
    __syncthreads();
    for (int i = tid; i < 1024; i += 256) {          // A: 128x32 f32 -> bf16
      int r = i >> 3;
      int c = (i & 7) << 2;
      float4 f = *(const float4*)(x + (m0 + r) * K + k0 + c);
      __bf16* d = &As[r * LDA + c];
      d[0] = (__bf16)f.x; d[1] = (__bf16)f.y; d[2] = (__bf16)f.z; d[3] = (__bf16)f.w;
    }
    for (int i = tid; i < 2048; i += 256) {          // B^T: Bs[n][c] = w[k0+c][n0+n]
      int n = i & 63;
      int c = i >> 6;
      Bs[n * LDA + c] = (__bf16)w[(size_t)(k0 + c) * N + n0 + n];
    }
    if (k0 + 32 < K)
      __builtin_prefetch(x + (m0 + (tid >> 1)) * K + k0 + 32, 0, 1);
    __syncthreads();

    FragBF a;
    frag_load(a, &As[(wave * 16 + l15) * LDA], khalf);
    FragBF b[4];
#pragma unroll
    for (int nt = 0; nt < 4; ++nt)
      frag_load(b[nt], &Bs[(nt * 16 + l15) * LDA], khalf);
#pragma unroll
    for (int nt = 0; nt < 4; ++nt)
      acc[nt] = wmma_bf16(a, b[nt], acc[nt]);
  }

#pragma unroll
  for (int nt = 0; nt < 4; ++nt) {
    int n     = n0 + nt * 16 + l15;
    int which = n / 768;
    int head  = (n % 768) >> 6;
    int c     = n & 63;
    __bf16* dst = (which == 0) ? qb : ((which == 1) ? kb : vb);
    float bv = bias[n];
#pragma unroll
    for (int i = 0; i < 8; ++i) {
      size_t m  = m0 + wave * 16 + rowoff + i;
      size_t bb = m / 196;
      size_t qq = m % 196;
      dst[((bb * 12 + head) * 196 + qq) * 64 + c] = (__bf16)(acc[nt][i] + bv);
    }
  }
}

// ---------------------------------------------------------------------------
// Kernel 2: fused attention per (b*head, 32-query block).
// Q tile arrives via TDM (tensor_load_to_lds) with OOB zero-fill + LDS padding.
// scores (WMMA) -> LDS; +scale +rel-pos bias; softmax; P@V (WMMA) -> bf16 out.
// ---------------------------------------------------------------------------
__global__ __launch_bounds__(256) void attn_kernel(
    const __bf16* __restrict__ qb, const __bf16* __restrict__ kb,
    const __bf16* __restrict__ vb, const float* __restrict__ rph,
    const float* __restrict__ rpw, __bf16* __restrict__ ob)
{
  __shared__ __bf16 Qs[32 * 72];     // 32 x 64, stride 72 (TDM pad: 32dw + 4dw)
  __shared__ float  Sc[32 * 208];
  __shared__ __bf16 Vt[64 * 224];    // V^T [c][key], zero-padded keys
  __shared__ float  Rh[32 * 16];
  __shared__ float  Rw[32 * 16];

  const int tid   = threadIdx.x;
  const int lane  = tid & 31;
  const int wave  = tid >> 5;
  const int l15   = lane & 15;
  const int khalf = (lane >> 4) << 3;
  const int rowoff = (lane >> 4) << 3;

  const int bh = blockIdx.y;
  const int q0 = blockIdx.x * 32;
  const __bf16* qp = qb + (size_t)bh * 196 * 64;
  const __bf16* kp = kb + (size_t)bh * 196 * 64;
  const __bf16* vp = vb + (size_t)bh * 196 * 64;

  // Q tile via Tensor Data Mover: tile 64x32 rows, tensor_dim1 = 196-q0 so rows
  // past the end zero-fill; LDS pad 4 dwords per 32-dword row -> stride 72.
  if (wave == 0) {
    tdm_load_2d(lds_offset_of(Qs), qp + (size_t)q0 * 64,
                /*tensor_d0=*/64, /*tensor_d1=*/(unsigned)(196 - q0),
                /*tile_d0=*/64, /*tile_d1=*/32,
                /*stride0=*/64, /*data_size=2B*/1,
                /*pad_interval 32dw*/4, /*pad_amount 4dw*/3, true);
  }
  // V^T staged manually (TDM has no transpose)
  for (int i = tid; i < 64 * 224; i += 256) {
    int c = i / 224, key = i % 224;
    Vt[c * 224 + key] = (key < 196) ? vp[(size_t)key * 64 + c] : (__bf16)0.0f;
  }
  if (wave == 0) __builtin_amdgcn_s_wait_tensorcnt(0);
  __syncthreads();

  // Decomposed rel-pos: Rh[r][kh] = sum_c q[r][c]*rel_pos_h[qh-kh+13][c]
  for (int idx = tid; idx < 32 * 14; idx += 256) {
    int r = idx / 14, kk = idx % 14;
    int qg = q0 + r;
    float sh = 0.f, sw = 0.f;
    if (qg < 196) {
      int qh = qg / 14, qw = qg % 14;
      const float* ph = rph + (size_t)(qh - kk + 13) * 64;
      const float* pw = rpw + (size_t)(qw - kk + 13) * 64;
      for (int c = 0; c < 64; ++c) {
        float qv = (float)Qs[r * 72 + c];
        sh += qv * ph[c];
        sw += qv * pw[c];
      }
    }
    Rh[r * 16 + kk] = sh;
    Rw[r * 16 + kk] = sw;
  }

  // Scores: 2(M) x 13(N) tiles, K=64 (2 WMMA steps); B frags straight from global k.
  for (int t = wave; t < 26; t += 8) {
    int mt = t / 13, nt = t % 13;
    v8f acc = {};
#pragma unroll
    for (int ks = 0; ks < 2; ++ks) {
      FragBF a, b;
      frag_load(a, &Qs[(mt * 16 + l15) * 72 + ks * 32], khalf);
      int key = nt * 16 + l15;
      if (key < 196) {
        frag_load(b, kp + (size_t)key * 64 + ks * 32, khalf);
      } else {
        b.q[0] = make_uint4(0, 0, 0, 0);
        b.q[1] = make_uint4(0, 0, 0, 0);
      }
      acc = wmma_bf16(a, b, acc);
    }
#pragma unroll
    for (int i = 0; i < 8; ++i)
      Sc[(mt * 16 + rowoff + i) * 208 + nt * 16 + l15] = acc[i];
  }
  __syncthreads();

  // scale + bias + softmax (rows are only 196 wide; one thread per row)
  if (tid < 32) {
    int qg = q0 + tid;
    if (qg < 196) {
      const float scale = 0.125f;
      float mx = -1e30f;
      for (int kh = 0; kh < 14; ++kh)
        for (int kw = 0; kw < 14; ++kw) {
          int kk = kh * 14 + kw;
          float s = Sc[tid * 208 + kk] * scale + Rh[tid * 16 + kh] + Rw[tid * 16 + kw];
          Sc[tid * 208 + kk] = s;
          mx = fmaxf(mx, s);
        }
      float sum = 0.f;
      for (int kk = 0; kk < 196; ++kk) {
        float e = __expf(Sc[tid * 208 + kk] - mx);
        Sc[tid * 208 + kk] = e;
        sum += e;
      }
      float inv = 1.0f / sum;
      for (int kk = 0; kk < 196; ++kk) Sc[tid * 208 + kk] *= inv;
    }
  }
  __syncthreads();

  // P @ V: 2(M) x 4(N) tiles, one per wave; K padded to 224 (7 steps)
  {
    int mt = wave >> 2, nt = wave & 3;
    v8f acc = {};
    for (int s = 0; s < 7; ++s) {
      FragBF a, b;
      int row = mt * 16 + l15;
#pragma unroll
      for (int j = 0; j < 4; ++j) {
        int kA = s * 32 + khalf + 2 * j;
        float f0 = (kA     < 196) ? Sc[row * 208 + kA]     : 0.f;
        float f1 = (kA + 1 < 196) ? Sc[row * 208 + kA + 1] : 0.f;
        a.u[j] = pack_bf2(f0, f1);
        int kB = kA + 16;
        f0 = (kB     < 196) ? Sc[row * 208 + kB]     : 0.f;
        f1 = (kB + 1 < 196) ? Sc[row * 208 + kB + 1] : 0.f;
        a.u[4 + j] = pack_bf2(f0, f1);
      }
      frag_load(b, &Vt[(nt * 16 + l15) * 224 + s * 32], khalf);
      acc = wmma_bf16(a, b, acc);
    }
    int bidx = bh / 12, head = bh % 12;
#pragma unroll
    for (int i = 0; i < 8; ++i) {
      int qg = q0 + mt * 16 + rowoff + i;
      if (qg < 196)
        ob[((size_t)(bidx * 196 + qg)) * 768 + head * 64 + nt * 16 + l15] =
            (__bf16)acc[i];
    }
  }
}

// ---------------------------------------------------------------------------
// Kernel 3: output projection.  a[50176,768] (bf16) @ w[768,768] (f32) + b -> f32
// A tile (already bf16) streamed by TDM each K-step with LDS padding;
// B tile converted manually.
// ---------------------------------------------------------------------------
__global__ __launch_bounds__(256) void proj_gemm_kernel(
    const __bf16* __restrict__ a_in, const float* __restrict__ w,
    const float* __restrict__ bias, float* __restrict__ out)
{
  constexpr int K = 768, N = 768, LDA = 40;
  __shared__ __bf16 As[128 * LDA];
  __shared__ __bf16 Bs[64 * LDA];

  const int tid   = threadIdx.x;
  const int lane  = tid & 31;
  const int wave  = tid >> 5;
  const int l15   = lane & 15;
  const int khalf = (lane >> 4) << 3;
  const int rowoff = (lane >> 4) << 3;

  const int    n0 = blockIdx.x * 64;
  const size_t m0 = (size_t)blockIdx.y * 128;

  v8f acc[4] = {};

  for (int k0 = 0; k0 < K; k0 += 32) {
    __syncthreads();   // previous compute done -> LDS safe to overwrite
    if (wave == 0) {
      // A tile: 128 rows x 32 bf16, row = 16 dwords + 4 dwords pad -> stride 40
      tdm_load_2d(lds_offset_of(As), a_in + m0 * K + k0,
                  /*tensor_d0=*/32, /*tensor_d1=*/128,
                  /*tile_d0=*/32, /*tile_d1=*/128,
                  /*stride0=*/(unsigned)K, /*data_size=2B*/1,
                  /*pad_interval 16dw*/3, /*pad_amount 4dw*/3, true);
    }
    for (int i = tid; i < 2048; i += 256) {          // B^T with f32->bf16
      int n = i & 63;
      int c = i >> 6;
      Bs[n * LDA + c] = (__bf16)w[(size_t)(k0 + c) * N + n0 + n];
    }
    if (k0 + 32 < K)
      __builtin_prefetch(a_in + (m0 + (tid >> 1)) * K + k0 + 32, 0, 1);
    if (wave == 0) __builtin_amdgcn_s_wait_tensorcnt(0);
    __syncthreads();

    FragBF a;
    frag_load(a, &As[(wave * 16 + l15) * LDA], khalf);
    FragBF b[4];
#pragma unroll
    for (int nt = 0; nt < 4; ++nt)
      frag_load(b[nt], &Bs[(nt * 16 + l15) * LDA], khalf);
#pragma unroll
    for (int nt = 0; nt < 4; ++nt)
      acc[nt] = wmma_bf16(a, b[nt], acc[nt]);
  }

#pragma unroll
  for (int nt = 0; nt < 4; ++nt) {
    int n = n0 + nt * 16 + l15;
    float bv = bias[n];
#pragma unroll
    for (int i = 0; i < 8; ++i) {
      size_t m = m0 + wave * 16 + rowoff + i;
      out[m * N + n] = acc[nt][i] + bv;
    }
  }
}

// ---------------------------------------------------------------------------
// Launch. Workspace (bf16): q,k,v [3072*196*64] + attnout [50176*768],
// each 38,535,168 elems -> 4 * 77,070,336 B = 308 MB.
// ---------------------------------------------------------------------------
extern "C" void kernel_launch(void* const* d_in, const int* in_sizes, int n_in,
                              void* d_out, int out_size, void* d_ws, size_t ws_size,
                              hipStream_t stream) {
  const float* x    = (const float*)d_in[0];
  const float* wqkv = (const float*)d_in[1];
  const float* bqkv = (const float*)d_in[2];
  const float* wprj = (const float*)d_in[3];
  const float* bprj = (const float*)d_in[4];
  const float* rph  = (const float*)d_in[5];
  const float* rpw  = (const float*)d_in[6];
  float* out = (float*)d_out;

  const size_t S = (size_t)3072 * 196 * 64;
  __bf16* qb = (__bf16*)d_ws;
  __bf16* kb = qb + S;
  __bf16* vb = kb + S;
  __bf16* ob = vb + S;

  qkv_gemm_kernel<<<dim3(2304 / 64, 50176 / 128), 256, 0, stream>>>(
      x, wqkv, bqkv, qb, kb, vb);
  attn_kernel<<<dim3(7, 3072), 256, 0, stream>>>(qb, kb, vb, rph, rpw, ob);
  proj_gemm_kernel<<<dim3(768 / 64, 50176 / 128), 256, 0, stream>>>(
      ob, wprj, bprj, out);
}
